// TransformerExpertBlock_51324859187254
// MI455X (gfx1250) — compile-verified, matrix operand order checked by reading, not run on hardware
//
#include <hip/hip_runtime.h>
#include <cstdint>

// ---------------------------------------------------------------------------
// Problem constants (from reference)
// ---------------------------------------------------------------------------
constexpr int D_   = 1024;
constexpr int C_   = 2048;
constexpr int MLP_ = 4096;
constexpr int H_   = 16;
constexpr int HD_  = 64;
constexpr int B_   = 8;
constexpr int N_   = 512;
constexpr int S_   = 2048;
constexpr int INNER_ = H_ * HD_;   // 1024
constexpr float EPS_ = 1e-6f;

typedef __attribute__((ext_vector_type(16))) __bf16 v16bf;
typedef __attribute__((ext_vector_type(8)))  float  v8f;

// ---------------------------------------------------------------------------
// gfx1250 async global->LDS copy helpers (ASYNCcnt tracked)
// ---------------------------------------------------------------------------
__device__ __forceinline__ void async_b128(unsigned lds_off, const __bf16* g) {
    asm volatile("global_load_async_to_lds_b128 %0, %1, off"
                 :: "v"(lds_off), "v"(g) : "memory");
}
__device__ __forceinline__ void wait_async0() {
    asm volatile("s_wait_asynccnt 0x0" ::: "memory");
}
__device__ __forceinline__ unsigned lds_off_of(const void* p) {
    return (unsigned)(uintptr_t)p;   // low 32 bits of generic LDS address
}

// ---------------------------------------------------------------------------
// mod = cond @ mw.T + mb    (B x 3D; tiny, scalar kernel, fp32)
// ---------------------------------------------------------------------------
__global__ void mod_kernel(const float* __restrict__ cond,
                           const float* __restrict__ mw,
                           const float* __restrict__ mb,
                           float* __restrict__ out) {
    int idx = blockIdx.x * blockDim.x + threadIdx.x;
    const int total = B_ * 3 * D_;
    if (idx >= total) return;
    int b = idx / (3 * D_);
    int j = idx - b * (3 * D_);
    const float* c = cond + (size_t)b * D_;
    const float* w = mw + (size_t)j * D_;
    float s = 0.f;
    #pragma unroll 4
    for (int k = 0; k < D_; ++k) s += c[k] * w[k];
    out[idx] = s + mb[j];
}

// ---------------------------------------------------------------------------
// fp32 -> bf16 bulk convert (weights)
// ---------------------------------------------------------------------------
__global__ void cvt_bf16_kernel(const float* __restrict__ s, __bf16* __restrict__ d, long n) {
    for (long i = blockIdx.x * 256L + threadIdx.x; i < n; i += (long)gridDim.x * 256)
        d[i] = (__bf16)s[i];
}

// ---------------------------------------------------------------------------
// adaRMS: normed(bf16) = x * rsqrt(mean(x^2)+eps) * w * (1+scale) + shift
// ---------------------------------------------------------------------------
__global__ void adarms_kernel(const float* __restrict__ x,
                              const float* __restrict__ w,
                              const float* __restrict__ mod,   // B x 3D
                              __bf16* __restrict__ normed) {
    __shared__ float red[256];
    const int row = blockIdx.x;             // b*N + n
    const int b   = row / N_;
    const float* xr = x + (size_t)row * D_;
    float s = 0.f;
    for (int d = threadIdx.x; d < D_; d += 256) { float v = xr[d]; s += v * v; }
    red[threadIdx.x] = s;
    __syncthreads();
    for (int off = 128; off > 0; off >>= 1) {
        if (threadIdx.x < off) red[threadIdx.x] += red[threadIdx.x + off];
        __syncthreads();
    }
    const float norm = rsqrtf(red[0] / (float)D_ + EPS_);
    const float* mo = mod + (size_t)b * 3 * D_;
    __bf16* nr = normed + (size_t)row * D_;
    for (int d = threadIdx.x; d < D_; d += 256) {
        float nv = xr[d] * norm * w[d];
        nr[d] = (__bf16)(nv * (1.f + mo[d]) + mo[D_ + d]);
    }
}

// ---------------------------------------------------------------------------
// Generic strided GEMM:  C[m,n] = alpha * sum_k A[m,k] * W[n,k]   (bf16 x bf16 -> f32)
//   A (bf16): A[m*lda + k]            (+ z*aBatch)   -- k contiguous
//   W (bf16): W[n*wns + k]            (+ z*wBatch)   -- k contiguous
//   out: addr = m*cRS + n*cCS         (+ z*cBatch)   -- to fp32 (Cf) or bf16 (Ch)
// M%128==0, K%32==0; N arbitrary (clamped loads, guarded stores).
// Double-buffered LDS, async global->LDS b128 staging.
// ---------------------------------------------------------------------------
constexpr int BM = 128, BN = 128, BK = 32;
constexpr int PADROW = 40;                 // padded LDS row: 32 bf16 + 16B pad (80B)

__global__ __launch_bounds__(256)
void gemm_bf16_kernel(const __bf16* __restrict__ A, long lda, long aBatch,
                      const __bf16* __restrict__ W, long wns, long wBatch,
                      float* Cf, __bf16* Ch, long cRS, long cCS, long cBatch,
                      int M, int N, int K, float alpha) {
    __shared__ __bf16 As[2][BM * PADROW];   // 2 x 10 KB
    __shared__ __bf16 Ws[2][BN * PADROW];   // 2 x 10 KB

    const __bf16* Ab = A + (size_t)blockIdx.z * aBatch;
    const __bf16* Wb = W + (size_t)blockIdx.z * wBatch;

    const int row0 = blockIdx.y * BM;
    const int col0 = blockIdx.x * BN;
    const int tid  = threadIdx.x;
    const int lane = tid & 31;
    const int wid  = tid >> 5;
    const int wm   = wid & 3;    // 4 wave rows  -> 32 rows each
    const int wn   = wid >> 2;   // 2 wave cols  -> 64 cols each

    // staging: tile = 128 rows x 32 bf16 = 512 x 16B chunks; 2 chunks/thread/operand
    const int rs  = tid >> 2;        // staging row (0..63), +64 for 2nd chunk
    const int seg = tid & 3;         // 16B segment within row
    const unsigned aOff[2] = { lds_off_of(&As[0][0]), lds_off_of(&As[1][0]) };
    const unsigned wOff[2] = { lds_off_of(&Ws[0][0]), lds_off_of(&Ws[1][0]) };

    auto stage = [&](int buf, int k0) {
        #pragma unroll
        for (int c = 0; c < 2; ++c) {
            int r = rs + c * 64;
            const __bf16* g = Ab + (size_t)(row0 + r) * lda + (k0 + seg * 8);
            async_b128(aOff[buf] + (unsigned)(r * PADROW + seg * 8) * 2, g);
        }
        #pragma unroll
        for (int c = 0; c < 2; ++c) {
            int r = rs + c * 64;
            int n = col0 + r; n = (n < N) ? n : (N - 1);
            const __bf16* g = Wb + (size_t)n * wns + (k0 + seg * 8);
            async_b128(wOff[buf] + (unsigned)(r * PADROW + seg * 8) * 2, g);
        }
    };

    v8f acc[2][4];
    #pragma unroll
    for (int mi = 0; mi < 2; ++mi)
        #pragma unroll
        for (int ni = 0; ni < 4; ++ni)
            acc[mi][ni] = (v8f){};

    const int nk = K / BK;
    stage(0, 0);
    wait_async0();
    __syncthreads();

    const int lr = lane & 15;           // row/col within 16
    const int kb = (lane >> 4) << 3;    // 0 or 8 (K-half select)

    for (int kt = 0; kt < nk; ++kt) {
        const int cur = kt & 1;
        if (kt + 1 < nk) stage(cur ^ 1, (kt + 1) * BK);

        const __bf16* AsC = As[cur];
        const __bf16* WsC = Ws[cur];

        v16bf afr[2], bfr[4];
        #pragma unroll
        for (int mi = 0; mi < 2; ++mi) {
            const __bf16* ap = AsC + (wm * 32 + mi * 16 + lr) * PADROW;
            #pragma unroll
            for (int j = 0; j < 8; ++j) {
                afr[mi][j]     = ap[kb + j];
                afr[mi][j + 8] = ap[16 + kb + j];
            }
        }
        #pragma unroll
        for (int ni = 0; ni < 4; ++ni) {
            const __bf16* bp = WsC + (wn * 64 + ni * 16 + lr) * PADROW;
            #pragma unroll
            for (int j = 0; j < 8; ++j) {
                bfr[ni][j]     = bp[kb + j];
                bfr[ni][j + 8] = bp[16 + kb + j];
            }
        }

        #pragma unroll
        for (int mi = 0; mi < 2; ++mi)
            #pragma unroll
            for (int ni = 0; ni < 4; ++ni)
                acc[mi][ni] = __builtin_amdgcn_wmma_f32_16x16x32_bf16(
                    false, afr[mi], false, bfr[ni],
                    (short)0, acc[mi][ni], false, false);

        if (kt + 1 < nk) wait_async0();
        __syncthreads();
    }

    // epilogue: C/D layout -- VGPR i: lanes 0-15 => M=i, lanes 16-31 => M=8+i
    float*  Cfb = Cf ? Cf + (size_t)blockIdx.z * cBatch : nullptr;
    __bf16* Chb = Ch ? Ch + (size_t)blockIdx.z * cBatch : nullptr;
    const int madd = (lane >> 4) << 3;
    #pragma unroll
    for (int mi = 0; mi < 2; ++mi) {
        #pragma unroll
        for (int ni = 0; ni < 4; ++ni) {
            int crow0 = row0 + wm * 32 + mi * 16 + madd;
            int ccol  = col0 + wn * 64 + ni * 16 + (lane & 15);
            if (ccol < N) {
                #pragma unroll
                for (int i = 0; i < 8; ++i) {
                    long addr = (long)(crow0 + i) * cRS + (long)ccol * cCS;
                    float v = alpha * acc[mi][ni][i];
                    if (Chb) Chb[addr] = (__bf16)v;
                    else     Cfb[addr] = v;
                }
            }
        }
    }
}

// ---------------------------------------------------------------------------
// row softmax: fp32 scores in -> bf16 probabilities out, one block per row
// ---------------------------------------------------------------------------
__global__ void softmax_kernel(const float* __restrict__ scores,
                               __bf16* __restrict__ probs, int L) {
    __shared__ float red[256];
    const float* r = scores + (size_t)blockIdx.x * L;
    __bf16*      o = probs  + (size_t)blockIdx.x * L;
    float mx = -3.4e38f;
    for (int i = threadIdx.x; i < L; i += 256) mx = fmaxf(mx, r[i]);
    red[threadIdx.x] = mx;
    __syncthreads();
    for (int off = 128; off > 0; off >>= 1) {
        if (threadIdx.x < off) red[threadIdx.x] = fmaxf(red[threadIdx.x], red[threadIdx.x + off]);
        __syncthreads();
    }
    mx = red[0];
    __syncthreads();
    float s = 0.f;
    for (int i = threadIdx.x; i < L; i += 256) s += __expf(r[i] - mx);
    red[threadIdx.x] = s;
    __syncthreads();
    for (int off = 128; off > 0; off >>= 1) {
        if (threadIdx.x < off) red[threadIdx.x] += red[threadIdx.x + off];
        __syncthreads();
    }
    float inv = 1.f / red[0];
    for (int i = threadIdx.x; i < L; i += 256) o[i] = (__bf16)(__expf(r[i] - mx) * inv);
}

// ---------------------------------------------------------------------------
// x[b,n,d] += gate[b,d] * y[b,n,d]   (gate = mod[:, 2D:3D]); fp32
// ---------------------------------------------------------------------------
__global__ void gated_residual_kernel(float* __restrict__ x,
                                      const float* __restrict__ y,
                                      const float* __restrict__ mod) {
    const long total = (long)B_ * N_ * D_;
    for (long i = blockIdx.x * 256L + threadIdx.x; i < total; i += (long)gridDim.x * 256) {
        int d = (int)(i % D_);
        int b = (int)(i / ((long)N_ * D_));
        x[i] += mod[(size_t)b * 3 * D_ + 2 * D_ + d] * y[i];
    }
}

// ---------------------------------------------------------------------------
// g = silu(g) * u   elementwise, bf16 in / bf16 out
// ---------------------------------------------------------------------------
__global__ void silu_mul_kernel(__bf16* __restrict__ g, const __bf16* __restrict__ u, long total) {
    for (long i = blockIdx.x * 256L + threadIdx.x; i < total; i += (long)gridDim.x * 256) {
        float gv = (float)g[i];
        float uv = (float)u[i];
        g[i] = (__bf16)(gv / (1.f + __expf(-gv)) * uv);
    }
}

// ---------------------------------------------------------------------------
// host orchestration
// ---------------------------------------------------------------------------
static inline void launch_gemm(hipStream_t st,
                               const __bf16* A, long lda, long aB,
                               const __bf16* W, long wns, long wB,
                               float* Cf, __bf16* Ch, long cRS, long cCS, long cB,
                               int M, int N, int K, int batch, float alpha) {
    dim3 grid((unsigned)((N + BN - 1) / BN), (unsigned)(M / BM), (unsigned)batch);
    gemm_bf16_kernel<<<grid, 256, 0, st>>>(A, lda, aB, W, wns, wB,
                                           Cf, Ch, cRS, cCS, cB, M, N, K, alpha);
}

extern "C" void kernel_launch(void* const* d_in, const int* in_sizes, int n_in,
                              void* d_out, int out_size, void* d_ws, size_t ws_size,
                              hipStream_t stream) {
    (void)in_sizes; (void)n_in; (void)out_size; (void)ws_size;

    const float* x     = (const float*)d_in[0];
    const float* ctx   = (const float*)d_in[1];
    const float* cond  = (const float*)d_in[2];
    const float* n1_w  = (const float*)d_in[3];
    const float* n1_mw = (const float*)d_in[4];
    const float* n1_mb = (const float*)d_in[5];
    const float* n2_w  = (const float*)d_in[6];
    const float* n2_mw = (const float*)d_in[7];
    const float* n2_mb = (const float*)d_in[8];
    const float* n3_w  = (const float*)d_in[9];
    const float* n3_mw = (const float*)d_in[10];
    const float* n3_mb = (const float*)d_in[11];
    const float* sa_q  = (const float*)d_in[12];
    const float* sa_k  = (const float*)d_in[13];
    const float* sa_v  = (const float*)d_in[14];
    const float* sa_o  = (const float*)d_in[15];
    const float* ca_q  = (const float*)d_in[16];
    const float* ca_k  = (const float*)d_in[17];
    const float* ca_v  = (const float*)d_in[18];
    const float* ca_o  = (const float*)d_in[19];
    const float* ff_g  = (const float*)d_in[20];
    const float* ff_u  = (const float*)d_in[21];
    const float* ff_d  = (const float*)d_in[22];

    float* out = (float*)d_out;

    // workspace carve-up (256B-aligned)
    uintptr_t wsp = (uintptr_t)d_ws;
    auto allocf = [&](size_t n) { wsp = (wsp + 255) & ~(uintptr_t)255; float*  p = (float*)wsp;  wsp += n * 4; return p; };
    auto allocb = [&](size_t n) { wsp = (wsp + 255) & ~(uintptr_t)255; __bf16* p = (__bf16*)wsp; wsp += n * 2; return p; };

    float*  mod1  = allocf((size_t)B_ * 3 * D_);
    float*  mod2  = allocf((size_t)B_ * 3 * D_);
    float*  mod3  = allocf((size_t)B_ * 3 * D_);
    float*  sbuf  = allocf((size_t)H_ * N_ * S_);        // per-batch scores (fp32)
    float*  pbuf  = allocf((size_t)B_ * N_ * D_);        // projection outputs (fp32)
    __bf16* normedb = allocb((size_t)B_ * N_ * D_);
    __bf16* qb    = allocb((size_t)B_ * N_ * INNER_);
    __bf16* kb    = allocb((size_t)B_ * S_ * INNER_);    // also FFN gate buffer
    __bf16* vtb   = allocb((size_t)B_ * S_ * INNER_);    // V^T per (b,h); also FFN up buffer
    __bf16* ab    = allocb((size_t)B_ * N_ * INNER_);    // attention output
    __bf16* pb    = allocb((size_t)H_ * N_ * S_);        // per-batch probabilities (bf16)
    // bf16 weights
    __bf16* wsaq = allocb((size_t)INNER_ * D_);
    __bf16* wsak = allocb((size_t)INNER_ * D_);
    __bf16* wsav = allocb((size_t)INNER_ * D_);
    __bf16* wsao = allocb((size_t)D_ * INNER_);
    __bf16* wcaq = allocb((size_t)INNER_ * D_);
    __bf16* wcak = allocb((size_t)INNER_ * C_);
    __bf16* wcav = allocb((size_t)INNER_ * C_);
    __bf16* wcao = allocb((size_t)D_ * INNER_);
    __bf16* wffg = allocb((size_t)MLP_ * D_);
    __bf16* wffu = allocb((size_t)MLP_ * D_);
    __bf16* wffd = allocb((size_t)D_ * MLP_);

    // weight conversion (once per launch)
    auto cvt = [&](const float* s, __bf16* d, long n) {
        cvt_bf16_kernel<<<2048, 256, 0, stream>>>(s, d, n);
    };
    cvt(sa_q, wsaq, (long)INNER_ * D_);
    cvt(sa_k, wsak, (long)INNER_ * D_);
    cvt(sa_v, wsav, (long)INNER_ * D_);
    cvt(sa_o, wsao, (long)D_ * INNER_);
    cvt(ca_q, wcaq, (long)INNER_ * D_);
    cvt(ca_k, wcak, (long)INNER_ * C_);
    cvt(ca_v, wcav, (long)INNER_ * C_);
    cvt(ca_o, wcao, (long)D_ * INNER_);
    cvt(ff_g, wffg, (long)MLP_ * D_);
    cvt(ff_u, wffu, (long)MLP_ * D_);
    cvt(ff_d, wffd, (long)D_ * MLP_);

    const int modBlocks = (B_ * 3 * D_ + 255) / 256;
    mod_kernel<<<modBlocks, 256, 0, stream>>>(cond, n1_mw, n1_mb, mod1);
    mod_kernel<<<modBlocks, 256, 0, stream>>>(cond, n2_mw, n2_mb, mod2);
    mod_kernel<<<modBlocks, 256, 0, stream>>>(cond, n3_mw, n3_mb, mod3);

    // residual accumulator lives in d_out
    hipMemcpyAsync(out, x, (size_t)B_ * N_ * D_ * sizeof(float),
                   hipMemcpyDeviceToDevice, stream);

    const float attnScale = 0.125f;   // HD^-0.5

    // ================= Self-attention (Ns = N_) =================
    adarms_kernel<<<B_ * N_, 256, 0, stream>>>(out, n1_w, mod1, normedb);
    // Q, K projections: normal bf16 output
    launch_gemm(stream, normedb, D_, 0, wsaq, D_, 0,
                nullptr, qb, INNER_, 1, 0, B_ * N_, INNER_, D_, 1, 1.f);
    launch_gemm(stream, normedb, D_, 0, wsak, D_, 0,
                nullptr, kb, INNER_, 1, 0, B_ * N_, INNER_, D_, 1, 1.f);
    // V projection, transposed per (b,h): VT[b][h][d][s], batch over b
    launch_gemm(stream, normedb, D_, (long)N_ * D_, wsav, D_, 0,
                nullptr, vtb, 1, N_, (long)INNER_ * N_, N_, INNER_, D_, B_, 1.f);
    for (int b = 0; b < B_; ++b) {
        const __bf16* qbb = qb  + (size_t)b * N_ * INNER_;
        const __bf16* kbb = kb  + (size_t)b * N_ * INNER_;
        const __bf16* vtbb= vtb + (size_t)b * INNER_ * N_;
        __bf16*       abb = ab  + (size_t)b * N_ * INNER_;
        // scores[h] = q_h @ k_h^T * scale   (N_ x N_), batch over H
        launch_gemm(stream, qbb, INNER_, HD_, kbb, INNER_, HD_,
                    sbuf, nullptr, N_, 1, (long)N_ * N_, N_, N_, HD_, H_, attnScale);
        softmax_kernel<<<H_ * N_, 256, 0, stream>>>(sbuf, pb, N_);
        // o_h = P_h @ V_h   (N_ x HD), W = VT_h (HD x N_), k-contiguous
        launch_gemm(stream, pb, N_, (long)N_ * N_, vtbb, N_, (long)HD_ * N_,
                    nullptr, abb, INNER_, 1, HD_, N_, HD_, N_, H_, 1.f);
    }
    launch_gemm(stream, ab, INNER_, 0, wsao, INNER_, 0,
                pbuf, nullptr, D_, 1, 0, B_ * N_, D_, INNER_, 1, 1.f);
    gated_residual_kernel<<<2048, 256, 0, stream>>>(out, pbuf, mod1);

    // ================= Cross-attention (Ns = S_) =================
    adarms_kernel<<<B_ * N_, 256, 0, stream>>>(out, n2_w, mod2, normedb);
    launch_gemm(stream, normedb, D_, 0, wcaq, D_, 0,
                nullptr, qb, INNER_, 1, 0, B_ * N_, INNER_, D_, 1, 1.f);
    // context K/V need bf16 context: convert once into pb buffer? No — convert into kb-sized temp.
    // Convert context to bf16 (reuse ab? too small). Use dedicated conversion into ctxb.
    // ctxb allocated below via static layout: reuse pb is too small (16M vs 32M ok actually:
    // ctx = B_*S_*C_ = 33.5M elems; pb is H_*N_*S_ = 16.7M -> NOT enough). Allocate properly:
    {
        static_assert((size_t)B_ * S_ * C_ == 33554432, "ctx size");
    }
    // (ctxb carved after weights; see below)
    // -- handled outside this comment block --

    // ================= placeholder to keep flow readable ================
    // (actual CA launches below after ctxb conversion)
    __bf16* ctxb = allocb((size_t)B_ * S_ * C_);
    cvt(ctx, ctxb, (long)B_ * S_ * C_);
    launch_gemm(stream, ctxb, C_, 0, wcak, C_, 0,
                nullptr, kb, INNER_, 1, 0, B_ * S_, INNER_, C_, 1, 1.f);
    launch_gemm(stream, ctxb, C_, (long)S_ * C_, wcav, C_, 0,
                nullptr, vtb, 1, S_, (long)INNER_ * S_, S_, INNER_, C_, B_, 1.f);
    for (int b = 0; b < B_; ++b) {
        const __bf16* qbb  = qb  + (size_t)b * N_ * INNER_;
        const __bf16* kbb  = kb  + (size_t)b * S_ * INNER_;
        const __bf16* vtbb = vtb + (size_t)b * INNER_ * S_;
        __bf16*       abb  = ab  + (size_t)b * N_ * INNER_;
        launch_gemm(stream, qbb, INNER_, HD_, kbb, INNER_, HD_,
                    sbuf, nullptr, S_, 1, (long)N_ * S_, N_, S_, HD_, H_, attnScale);
        softmax_kernel<<<H_ * N_, 256, 0, stream>>>(sbuf, pb, S_);
        launch_gemm(stream, pb, S_, (long)N_ * S_, vtbb, S_, (long)HD_ * S_,
                    nullptr, abb, INNER_, 1, HD_, N_, HD_, S_, H_, 1.f);
    }
    launch_gemm(stream, ab, INNER_, 0, wcao, INNER_, 0,
                pbuf, nullptr, D_, 1, 0, B_ * N_, D_, INNER_, 1, 1.f);
    gated_residual_kernel<<<2048, 256, 0, stream>>>(out, pbuf, mod2);

    // ================= SwiGLU FFN =================
    adarms_kernel<<<B_ * N_, 256, 0, stream>>>(out, n3_w, mod3, normedb);
    launch_gemm(stream, normedb, D_, 0, wffg, D_, 0,
                nullptr, kb, MLP_, 1, 0, B_ * N_, MLP_, D_, 1, 1.f);
    launch_gemm(stream, normedb, D_, 0, wffu, D_, 0,
                nullptr, vtb, MLP_, 1, 0, B_ * N_, MLP_, D_, 1, 1.f);
    silu_mul_kernel<<<2048, 256, 0, stream>>>(kb, vtb, (long)B_ * N_ * MLP_);
    launch_gemm(stream, kb, MLP_, 0, wffd, MLP_, 0,
                pbuf, nullptr, D_, 1, 0, B_ * N_, D_, MLP_, 1, 1.f);
    gated_residual_kernel<<<2048, 256, 0, stream>>>(out, pbuf, mod3);
}
